// BiBoAttention_15333033247085
// MI455X (gfx1250) — compile-verified, matrix operand order checked by reading, not run on hardware
//
#include <hip/hip_runtime.h>
#include <hip/hip_bf16.h>

// ---------------------------------------------------------------------------
// Problem constants (from reference)
// ---------------------------------------------------------------------------
constexpr int BB  = 2;
constexpr int SS  = 2048;
constexpr int HID = 1024;
constexpr int NH  = 16;
constexpr int NKV = 4;
constexpr int HD  = 64;
constexpr float EPS = 1e-6f;
constexpr int MROWS = BB * SS;          // 4096 flattened rows

typedef __attribute__((ext_vector_type(16))) _Float16 v16h;
typedef __attribute__((ext_vector_type(8)))  _Float16 v8h;
typedef __attribute__((ext_vector_type(8)))  float    v8f;
typedef __attribute__((ext_vector_type(4)))  int      v4i;

// ---------------------------------------------------------------------------
// Async global->LDS copy (CDNA5 GLOBAL_LOAD_ASYNC_TO_LDS_B128, ASYNCcnt)
// Builtin signature (from hipcc diagnostic): (v4i32 AS1*, v4i32 AS3*, int, int)
// ---------------------------------------------------------------------------
#if defined(__gfx1250__) && __has_builtin(__builtin_amdgcn_global_load_async_to_lds_b128)
#define HAVE_ASYNC_LDS 1
typedef __attribute__((address_space(1))) v4i gv4i;
typedef __attribute__((address_space(3))) v4i lv4i;
#endif

__device__ __forceinline__ void async_copy_b128(_Float16* lds_dst,
                                                const _Float16* gsrc) {
#if defined(HAVE_ASYNC_LDS)
    __builtin_amdgcn_global_load_async_to_lds_b128(
        (gv4i*)gsrc, (lv4i*)lds_dst, 0, 0);
#else
    *(v8h*)lds_dst = *(const v8h*)gsrc;
#endif
}

__device__ __forceinline__ void async_copy_wait() {
#if defined(HAVE_ASYNC_LDS)
#if __has_builtin(__builtin_amdgcn_s_wait_asynccnt)
    __builtin_amdgcn_s_wait_asynccnt(0);
#else
    asm volatile("s_wait_asynccnt 0x0" ::: "memory");
#endif
#endif
}

// ---------------------------------------------------------------------------
// WMMA fragment helpers (CDNA5 wave32 VGPR layouts, ISA 7.12.2)
//   A (16x32 f16): lane m = lane&15, lane half g = lane>>4
//     vgpr v<4  -> K = 2v + 8g ;  vgpr v>=4 -> K = 16 + 2(v-4) + 8g
//   B/C/D: col n = lane&15, row m = r + 8g
// ---------------------------------------------------------------------------
__device__ __forceinline__ int frag_k(int v, int g) {
    return (v < 4) ? (2 * v + 8 * g) : (16 + 2 * (v - 4) + 8 * g);
}

// A fragment from row-major 16xK tile (stride lda, f16 elements)
__device__ __forceinline__ v16h load_frag_a(const _Float16* src, int lda) {
    int lane = threadIdx.x & 31;
    int m = lane & 15, g = lane >> 4;
    v16h a;
#pragma unroll
    for (int v = 0; v < 8; ++v) {
        int k = frag_k(v, g);
        a[2 * v]     = src[m * lda + k];
        a[2 * v + 1] = src[m * lda + k + 1];
    }
    return a;
}

// B fragment where GEMM-B[k][n] = src[n*ldb + k]   (N x K row-major)
// Contiguous along k -> vectorizes to ds_load_b128 pairs.
__device__ __forceinline__ v16h load_frag_b_nk(const _Float16* src, int ldb) {
    int lane = threadIdx.x & 31;
    int n = lane & 15, g = lane >> 4;
    v16h b;
#pragma unroll
    for (int v = 0; v < 8; ++v) {
        int k = frag_k(v, g);
        b[2 * v]     = src[n * ldb + k];
        b[2 * v + 1] = src[n * ldb + k + 1];
    }
    return b;
}

__device__ __forceinline__ void store_frag_c(float* dst, int ldc, v8f c) {
    int lane = threadIdx.x & 31;
    int n = lane & 15, g = lane >> 4;
#pragma unroll
    for (int r = 0; r < 8; ++r) dst[(r + 8 * g) * ldc + n] = c[r];
}

__device__ __forceinline__ v8f wmma_f16(v16h a, v16h b, v8f c) {
    return __builtin_amdgcn_wmma_f32_16x16x32_f16(false, a, false, b, (short)0, c,
                                                  false, false);
}

// ---------------------------------------------------------------------------
// f32 -> f16 convert
// ---------------------------------------------------------------------------
__global__ __launch_bounds__(256) void cvt_f32_f16(const float* __restrict__ in,
                                                   _Float16* __restrict__ out, int n) {
    int i = blockIdx.x * 256 + threadIdx.x;
    if (i < n) out[i] = (_Float16)in[i];
}

// ---------------------------------------------------------------------------
// Tiled WMMA GEMM: C[M,N](f32) = A[M,K](f16, row-major) @ W[N,K](f16)^T
// Block tile 128x64, 8 waves as 4(M) x 2(N), each wave 32x32 (2x2 frags)
// ---------------------------------------------------------------------------
constexpr int TBM = 128;
constexpr int TBN = 64;
constexpr int TBK = 32;

__global__ __launch_bounds__(256) void gemm_f16_f32(
    const _Float16* __restrict__ A, const _Float16* __restrict__ W,
    float* __restrict__ C, int M, int N, int K)
{
    __shared__ _Float16 lA[TBM * TBK];   // 8 KB
    __shared__ _Float16 lB[TBN * TBK];   // 4 KB
    const int m0 = blockIdx.x * TBM, n0 = blockIdx.y * TBN;
    const int t = threadIdx.x;
    const int wave = t >> 5;
    const int wm = wave & 3, wn = wave >> 2;

    // staging coordinates (three 16B segments per thread)
    const int rowA0 = t >> 2, segA0 = t & 3;            // A segs 0..255
    const int rowA1 = (t + 256) >> 2, segA1 = t & 3;    // A segs 256..511
    const int rowB = t >> 2, segB = t & 3;              // W segs 0..255

    v8f c[2][2] = {};

    for (int k0 = 0; k0 < K; k0 += TBK) {
        const _Float16* gA0 = &A[(size_t)(m0 + rowA0) * K + k0 + segA0 * 8];
        const _Float16* gA1 = &A[(size_t)(m0 + rowA1) * K + k0 + segA1 * 8];
        const _Float16* gB  = &W[(size_t)(n0 + rowB) * K + k0 + segB * 8];
#if defined(HAVE_ASYNC_LDS)
        async_copy_b128(&lA[rowA0 * TBK + segA0 * 8], gA0);
        async_copy_b128(&lA[rowA1 * TBK + segA1 * 8], gA1);
        async_copy_b128(&lB[rowB * TBK + segB * 8], gB);
        async_copy_wait();
#else
        // issue all global loads before any LDS store (keeps loads in flight)
        v8h tA0 = *(const v8h*)gA0;
        v8h tA1 = *(const v8h*)gA1;
        v8h tB  = *(const v8h*)gB;
        *(v8h*)&lA[rowA0 * TBK + segA0 * 8] = tA0;
        *(v8h*)&lA[rowA1 * TBK + segA1 * 8] = tA1;
        *(v8h*)&lB[rowB * TBK + segB * 8]   = tB;
#endif
        // prefetch next K slab while WMMA runs
        if (k0 + TBK < K) {
            __builtin_prefetch(&A[(size_t)(m0 + (t >> 1)) * K + k0 + TBK], 0, 1);
            __builtin_prefetch(&W[(size_t)(n0 + (t & 63)) * K + k0 + TBK], 0, 1);
        }
        __syncthreads();

#pragma unroll
        for (int i = 0; i < 2; ++i) {
            v16h a = load_frag_a(&lA[(wm * 32 + i * 16) * TBK], TBK);
#pragma unroll
            for (int j = 0; j < 2; ++j) {
                v16h b = load_frag_b_nk(&lB[(wn * 32 + j * 16) * TBK], TBK);
                c[i][j] = wmma_f16(a, b, c[i][j]);
            }
        }
        __syncthreads();
    }

#pragma unroll
    for (int i = 0; i < 2; ++i)
#pragma unroll
        for (int j = 0; j < 2; ++j)
            store_frag_c(&C[(size_t)(m0 + wm * 32 + i * 16) * N + n0 + wn * 32 + j * 16],
                         N, c[i][j]);
}

// ---------------------------------------------------------------------------
// Per-head RMSNorm + RoPE + GQA relayout (one wave32 per (b,s,unit))
// units: [0,NH) = Q heads, [NH,NH+NKV) = K heads, [NH+NKV,NH+2NKV) = V heads
// Emits f16 [B, H, S, HD]; Q pre-scaled by 1/sqrt(HD).
// ---------------------------------------------------------------------------
__global__ __launch_bounds__(256) void norm_rope_relayout(
    const float* __restrict__ Q0, const float* __restrict__ K0,
    const float* __restrict__ V0, const float* __restrict__ cosT,
    const float* __restrict__ sinT, const int* __restrict__ pos_ids,
    const float* __restrict__ qw, const float* __restrict__ kw,
    _Float16* __restrict__ Qh, _Float16* __restrict__ Kh, _Float16* __restrict__ Vh)
{
    constexpr int UNITS = NH + 2 * NKV;   // 24
    int wg = blockIdx.x * 8 + (threadIdx.x >> 5);
    int lane = threadIdx.x & 31;
    int u = wg % UNITS;
    int s = (wg / UNITS) % SS;
    int b = wg / (UNITS * SS);

    const float* src;
    _Float16* dst;
    const float* w = nullptr;
    bool rope = true;
    float post = 1.0f;

    if (u < NH) {                                   // Q
        src = Q0 + ((size_t)(b * SS + s) * NH + u) * HD;
        dst = Qh + (((size_t)(b * NH + u)) * SS + s) * HD;
        w = qw; post = 0.125f;                      // 1/sqrt(64)
    } else if (u < NH + NKV) {                      // K
        int hh = u - NH;
        src = K0 + ((size_t)(b * SS + s) * NKV + hh) * HD;
        dst = Kh + (((size_t)(b * NKV + hh)) * SS + s) * HD;
        w = kw;
    } else {                                        // V: convert + relayout only
        int hh = u - NH - NKV;
        src = V0 + ((size_t)(b * SS + s) * NKV + hh) * HD;
        dst = Vh + (((size_t)(b * NKV + hh)) * SS + s) * HD;
        rope = false;
    }

    float x0 = src[lane];
    float x1 = src[lane + 32];

    if (rope) {
        float ssum = x0 * x0 + x1 * x1;
#pragma unroll
        for (int off = 16; off > 0; off >>= 1) ssum += __shfl_xor(ssum, off, 32);
        float inv = rsqrtf(ssum * (1.0f / HD) + EPS);
        x0 = x0 * inv * w[lane];
        x1 = x1 * inv * w[lane + 32];
        int pos = pos_ids[b * SS + s];
        float c0 = cosT[(size_t)pos * HD + lane];
        float c1 = cosT[(size_t)pos * HD + lane + 32];
        float s0 = sinT[(size_t)pos * HD + lane];
        float s1 = sinT[(size_t)pos * HD + lane + 32];
        float r0 = x0 * c0 - x1 * s0;   // rotate_half: first half gets -x2*sin
        float r1 = x1 * c1 + x0 * s1;   // second half gets +x1*sin
        x0 = r0 * post;
        x1 = r1 * post;
    }
    dst[lane]      = (_Float16)x0;
    dst[lane + 32] = (_Float16)x1;
}

// ---------------------------------------------------------------------------
// Flash attention: grid (S/64, NH, B), 128 threads = 4 waves.
// Each wave owns a 16-row Q strip; online softmax in the WMMA C layout;
// P re-shaped C->A layout through per-wave LDS staging.
// V is staged TRANSPOSED (Vt[d][key]) so P*V B-fragments are contiguous.
// ---------------------------------------------------------------------------
__global__ __launch_bounds__(128) void flash_attn(
    const _Float16* __restrict__ Qh, const _Float16* __restrict__ Kh,
    const _Float16* __restrict__ Vh, _Float16* __restrict__ Ctx)
{
    __shared__ _Float16 Qs[64 * 64];       // 8 KB, [qrow][d]
    __shared__ _Float16 Ks[64 * 64];       // 8 KB, [key][d]
    __shared__ _Float16 Vt[64 * 64];       // 8 KB, [d][key]  (transposed V)
    __shared__ _Float16 Ps[4][16 * 64];    // 8 KB, per-wave P staging

    const int qt = blockIdx.x, h = blockIdx.y, b = blockIdx.z;
    const int hk = h / (NH / NKV);
    const int t = threadIdx.x, w = t >> 5, lane = t & 31;
    const int ln = lane & 15, g = lane >> 4;

    const _Float16* Qg = Qh + (((size_t)(b * NH + h)) * SS + qt * 64) * HD;
    const _Float16* Kg = Kh + (((size_t)(b * NKV + hk)) * SS) * HD;
    const _Float16* Vg = Vh + (((size_t)(b * NKV + hk)) * SS) * HD;

#pragma unroll
    for (int i = 0; i < 4; ++i) {
        int idx = t + i * 128;             // 512 x v8h
        async_copy_b128(&Qs[idx * 8], &Qg[idx * 8]);
    }
    async_copy_wait();
    __syncthreads();

    v8f acc[4] = {};
    float mrow[8], lrow[8];
#pragma unroll
    for (int r = 0; r < 8; ++r) { mrow[r] = -1e30f; lrow[r] = 0.f; }

    for (int kb = 0; kb <= qt; ++kb) {
        // K tile: direct (async) copy, [key][d]
#pragma unroll
        for (int i = 0; i < 4; ++i) {
            int idx = t + i * 128;
            async_copy_b128(&Ks[idx * 8], &Kg[(size_t)kb * 64 * HD + idx * 8]);
        }
        // V tile: transpose into Vt[d][key] via registers
#pragma unroll
        for (int i = 0; i < 4; ++i) {
            int idx = t + i * 128;
            int key = idx >> 3, seg = idx & 7;
            v8h vv = *(const v8h*)&Vg[(size_t)kb * 64 * HD + key * HD + seg * 8];
#pragma unroll
            for (int e = 0; e < 8; ++e) Vt[(seg * 8 + e) * 64 + key] = vv[e];
        }
        async_copy_wait();
        __syncthreads();

        // S = Q * K^T  (16x64 per wave, 4 col tiles, K-dim = HD = 64)
        v8f sf[4];
#pragma unroll
        for (int j = 0; j < 4; ++j) {
            v8f sc = {};
#pragma unroll
            for (int kc = 0; kc < HD; kc += 32) {
                v16h a = load_frag_a(&Qs[(w * 16) * HD + kc], HD);
                v16h bm = load_frag_b_nk(&Ks[(j * 16) * HD + kc], HD);  // B[d][key]=K[key][d]
                sc = wmma_f16(a, bm, sc);
            }
            sf[j] = sc;
        }

        const bool diag = (kb == qt);
        float newm[8];
#pragma unroll
        for (int r = 0; r < 8; ++r) {
            int qrow = qt * 64 + w * 16 + r + 8 * g;
            float mx = mrow[r];
#pragma unroll
            for (int j = 0; j < 4; ++j) {
                int col = kb * 64 + j * 16 + ln;
                float v = sf[j][r];
                if (diag && col > qrow) { v = -1e30f; sf[j][r] = v; }
                mx = fmaxf(mx, v);
            }
#pragma unroll
            for (int off = 8; off > 0; off >>= 1)
                mx = fmaxf(mx, __shfl_xor(mx, off, 32));
            newm[r] = mx;
        }

#pragma unroll
        for (int r = 0; r < 8; ++r) {
            float scale = __expf(mrow[r] - newm[r]);
            float rs = 0.f;
#pragma unroll
            for (int j = 0; j < 4; ++j) {
                float p = __expf(sf[j][r] - newm[r]);
                rs += p;
                Ps[w][(r + 8 * g) * 64 + j * 16 + ln] = (_Float16)p;
            }
#pragma unroll
            for (int off = 8; off > 0; off >>= 1) rs += __shfl_xor(rs, off, 32);
            lrow[r] = lrow[r] * scale + rs;
            mrow[r] = newm[r];
#pragma unroll
            for (int j = 0; j < 4; ++j) acc[j][r] *= scale;
        }

        // O += P * V   (K-dim = 64 keys); B[key][d] = Vt[d][key] -> _nk loader
#pragma unroll
        for (int j = 0; j < 4; ++j) {
#pragma unroll
            for (int kc = 0; kc < 64; kc += 32) {
                v16h a  = load_frag_a(&Ps[w][kc], 64);
                v16h bm = load_frag_b_nk(&Vt[(j * 16) * 64 + kc], 64);
                acc[j] = wmma_f16(a, bm, acc[j]);
            }
        }
        __syncthreads();
    }

    // normalize + write ctx as f16 [B, S, NH*HD]
#pragma unroll
    for (int j = 0; j < 4; ++j) {
#pragma unroll
        for (int r = 0; r < 8; ++r) {
            int row = qt * 64 + w * 16 + r + 8 * g;
            float o = acc[j][r] / lrow[r];
            Ctx[((size_t)(b * SS + row)) * (NH * HD) + h * HD + j * 16 + ln] =
                (_Float16)o;
        }
    }
}

// ---------------------------------------------------------------------------
// Host-side launch
// ---------------------------------------------------------------------------
extern "C" void kernel_launch(void* const* d_in, const int* in_sizes, int n_in,
                              void* d_out, int out_size, void* d_ws, size_t ws_size,
                              hipStream_t stream)
{
    const float* hidden = (const float*)d_in[0];
    const float* cosT   = (const float*)d_in[1];
    const float* sinT   = (const float*)d_in[2];
    const int*   pos    = (const int*)d_in[3];
    // d_in[4] = attention_mask: causal mask applied analytically in flash_attn
    const float* Wq = (const float*)d_in[5];
    const float* Wk = (const float*)d_in[6];
    const float* Wv = (const float*)d_in[7];
    const float* Wo = (const float*)d_in[8];
    const float* qw = (const float*)d_in[9];
    const float* kw = (const float*)d_in[10];
    float* out = (float*)d_out;

    char* ws = (char*)d_ws;
    size_t off = 0;
    auto alloc = [&](size_t bytes) -> void* {
        void* p = ws + off;
        off = (off + bytes + 255) & ~(size_t)255;
        return p;
    };

    _Float16* h16  = (_Float16*)alloc((size_t)MROWS * HID * 2);
    _Float16* wq16 = (_Float16*)alloc((size_t)NH * HD * HID * 2);
    _Float16* wk16 = (_Float16*)alloc((size_t)NKV * HD * HID * 2);
    _Float16* wv16 = (_Float16*)alloc((size_t)NKV * HD * HID * 2);
    _Float16* wo16 = (_Float16*)alloc((size_t)HID * NH * HD * 2);
    float*    q0   = (float*)alloc((size_t)MROWS * NH * HD * 4);
    float*    k0   = (float*)alloc((size_t)MROWS * NKV * HD * 4);
    float*    v0   = (float*)alloc((size_t)MROWS * NKV * HD * 4);
    _Float16* qh   = (_Float16*)alloc((size_t)MROWS * NH * HD * 2);
    _Float16* kh   = (_Float16*)alloc((size_t)MROWS * NKV * HD * 2);
    _Float16* vh   = (_Float16*)alloc((size_t)MROWS * NKV * HD * 2);
    _Float16* ctx  = (_Float16*)alloc((size_t)MROWS * NH * HD * 2);

    // 1) f32 -> f16 conversions
    auto cvt = [&](const float* src, _Float16* dst, int n) {
        cvt_f32_f16<<<(n + 255) / 256, 256, 0, stream>>>(src, dst, n);
    };
    cvt(hidden, h16, MROWS * HID);
    cvt(Wq, wq16, NH * HD * HID);
    cvt(Wk, wk16, NKV * HD * HID);
    cvt(Wv, wv16, NKV * HD * HID);
    cvt(Wo, wo16, HID * NH * HD);

    // 2) QKV projections (WMMA GEMM)
    gemm_f16_f32<<<dim3(MROWS / TBM, (NH * HD) / TBN), 256, 0, stream>>>(
        h16, wq16, q0, MROWS, NH * HD, HID);
    gemm_f16_f32<<<dim3(MROWS / TBM, (NKV * HD) / TBN), 256, 0, stream>>>(
        h16, wk16, k0, MROWS, NKV * HD, HID);
    gemm_f16_f32<<<dim3(MROWS / TBM, (NKV * HD) / TBN), 256, 0, stream>>>(
        h16, wv16, v0, MROWS, NKV * HD, HID);

    // 3) RMSNorm + RoPE + relayout (one wave per (b,s,head-unit))
    {
        int waves = BB * SS * (NH + 2 * NKV);
        norm_rope_relayout<<<waves / 8, 256, 0, stream>>>(
            q0, k0, v0, cosT, sinT, pos, qw, kw, qh, kh, vh);
    }

    // 4) Flash attention (WMMA QK^T and PV)
    flash_attn<<<dim3(SS / 64, NH, BB), 128, 0, stream>>>(qh, kh, vh, ctx);

    // 5) Output projection -> f32 d_out
    gemm_f16_f32<<<dim3(MROWS / TBM, HID / TBN), 256, 0, stream>>>(
        ctx, wo16, out, MROWS, HID, NH * HD);
}
